// AnomalyAttention_68599217652261
// MI455X (gfx1250) — matrix operand order repaired and checked by reading, not compiled, not run on hardware
//
#include <hip/hip_runtime.h>

#define B_ 4
#define L_ 2048
#define H_ 8
#define E_ 64
// fold softmax scale (1/sqrt(64)) and log2(e) into the Q->f16 conversion:
// scores come out of WMMA already in the exp2 domain.
#define QSCALE 0.18033688011112042f  // 0.125 * 1.4426950408889634

typedef __attribute__((ext_vector_type(16))) _Float16 v16h;
typedef __attribute__((ext_vector_type(8)))  _Float16 v8h;
typedef __attribute__((ext_vector_type(2)))  _Float16 v2h;
typedef __attribute__((ext_vector_type(8)))  float    v8f;
typedef __attribute__((ext_vector_type(4)))  float    v4f;

// LDS row strides (in halves), padded for bank-conflict-free b128 reads
#define QSTR 72
#define KSTR 72
#define VSTR 40
#define PSTR 40

__device__ __forceinline__ v8f wmma16(v16h a, v16h b, v8f c) {
  return __builtin_amdgcn_wmma_f32_16x16x32_f16(false, a, false, b, (short)0, c,
                                                false, false);
}

__device__ __forceinline__ v16h cat16(v8h lo, v8h hi) {
  return __builtin_shufflevector(lo, hi, 0, 1, 2, 3, 4, 5, 6, 7,
                                 8, 9, 10, 11, 12, 13, 14, 15);
}

__device__ __forceinline__ float fexp2(float x) {
#if __has_builtin(__builtin_amdgcn_exp2f)
  return __builtin_amdgcn_exp2f(x);  // v_exp_f32
#else
  return exp2f(x);
#endif
}

// raw max without fmaxf's canonicalization (no NaNs possible in this kernel)
__device__ __forceinline__ float rawmax(float a, float b) {
  float r;
  asm("v_max_num_f32 %0, %1, %2" : "=v"(r) : "v"(a), "v"(b));
  return r;
}

// one-instruction butterfly stages: max with the DPP8-permuted self
__device__ __forceinline__ float maxx1(float x) {
  float r;
  asm("v_max_num_f32 %0, %1, %2 dpp8:[1,0,3,2,5,4,7,6]" : "=v"(r) : "v"(x), "v"(x));
  return r;
}
__device__ __forceinline__ float maxx2(float x) {
  float r;
  asm("v_max_num_f32 %0, %1, %2 dpp8:[2,3,0,1,6,7,4,5]" : "=v"(r) : "v"(x), "v"(x));
  return r;
}
__device__ __forceinline__ float maxx4(float x) {
  float r;
  asm("v_max_num_f32 %0, %1, %2 dpp8:[4,5,6,7,0,1,2,3]" : "=v"(r) : "v"(x), "v"(x));
  return r;
}

// lane ^ 8 within each 16-lane row: VALU permlane if available, else ds_swizzle
__device__ __forceinline__ float lanexor8(float x) {
  int xi = __builtin_bit_cast(int, x);
#if __has_builtin(__builtin_amdgcn_permlane16)
  int y = __builtin_amdgcn_permlane16(xi, xi, 0xFEDCBA98u, 0x76543210u, false, false);
#else
  int y = __builtin_amdgcn_ds_swizzle(xi, 0x201F);  // group-of-32: xor=8, and=0x1F
#endif
  return __builtin_bit_cast(float, y);
}

// max over the 16-lane half of the wave (halves stay independent)
__device__ __forceinline__ float rowmax16(float x) {
  x = maxx1(x);
  x = maxx2(x);
  x = maxx4(x);
  return rawmax(x, lanexor8(x));
}

// 8 floats -> v8h via packed RTZ converts
__device__ __forceinline__ v8h pk8(v4f a, v4f b) {
  union {
    v8h v;
    v2h p[4];
  } u;
  u.p[0] = __builtin_bit_cast(v2h, __builtin_amdgcn_cvt_pkrtz(a[0], a[1]));
  u.p[1] = __builtin_bit_cast(v2h, __builtin_amdgcn_cvt_pkrtz(a[2], a[3]));
  u.p[2] = __builtin_bit_cast(v2h, __builtin_amdgcn_cvt_pkrtz(b[0], b[1]));
  u.p[3] = __builtin_bit_cast(v2h, __builtin_amdgcn_cvt_pkrtz(b[2], b[3]));
  return u.v;
}

// per-thread register staging of the next K/V tile (16+16 floats)
struct Stage {
  v4f k0, k1, k2, k3;
  v4f v0, v1, v2, v3;
};

__global__ __launch_bounds__(128) void flash_attn_f16wmma(
    const float* __restrict__ Q, const float* __restrict__ K,
    const float* __restrict__ V, float* __restrict__ Out) {
  __shared__ __align__(16) _Float16 qlds[64 * QSTR];      // 64x64 Q (f16, pre-scaled)
  __shared__ __align__(16) _Float16 klds[2][32 * KSTR];   // K tiles, double buffered
  __shared__ __align__(16) _Float16 vtlds[2][64 * VSTR];  // V^T tiles [d][perm(key)]
  __shared__ __align__(16) _Float16 plds[4 * 16 * PSTR];  // per-wave P scratch 16x32

  const int tid  = threadIdx.x;
  const int lane = tid & 31;
  const int wv   = tid >> 5;   // wave 0..3
  const int ln   = lane & 15;  // lane within half
  const int lh   = lane >> 4;  // half: 0 or 1

  const int q0 = blockIdx.x * 64;
  const int bh = blockIdx.y;
  const int b  = bh >> 3;
  const int h  = bh & 7;

  const size_t rowstride = (size_t)H_ * E_;  // 512 floats between seq positions
  const float* qbase = Q + (size_t)b * L_ * rowstride + (size_t)h * E_;
  const float* kbase = K + (size_t)b * L_ * rowstride + (size_t)h * E_;
  const float* vbase = V + (size_t)b * L_ * rowstride + (size_t)h * E_;
  float*       obase = Out + (size_t)b * L_ * rowstride + (size_t)h * E_;

  // staging thread mapping for K/V tiles: key row = tid>>2, cols = (tid&3)*16
  const int sr  = tid >> 2;
  const int sc0 = (tid & 3) * 16;
  // key-permuted slot in V^T: score-tile0 keys -> even cols, tile1 -> odd cols
  const int sp = ((sr & 15) << 1) | (sr >> 4);

  // ---- stage Q tile (64x64), folding QSCALE into the f16 conversion ----
  {
    int r  = tid >> 1;
    int c0 = (tid & 1) * 32;
    const float* g = qbase + (size_t)(q0 + r) * rowstride + c0;
#pragma unroll
    for (int i = 0; i < 2; ++i) {
      v4f a0 = *(const v4f*)(g + i * 16)      * QSCALE;
      v4f a1 = *(const v4f*)(g + i * 16 + 4)  * QSCALE;
      v4f a2 = *(const v4f*)(g + i * 16 + 8)  * QSCALE;
      v4f a3 = *(const v4f*)(g + i * 16 + 12) * QSCALE;
      *(v8h*)&qlds[r * QSTR + c0 + i * 16]     = pk8(a0, a1);
      *(v8h*)&qlds[r * QSTR + c0 + i * 16 + 8] = pk8(a2, a3);
    }
  }

  // ---- stage tile 0 of K/V into buffer 0 ----
  {
    const float* gk = kbase + (size_t)sr * rowstride + sc0;
    const float* gv = vbase + (size_t)sr * rowstride + sc0;
    v4f k0 = *(const v4f*)gk, k1 = *(const v4f*)(gk + 4);
    v4f k2 = *(const v4f*)(gk + 8), k3 = *(const v4f*)(gk + 12);
    *(v8h*)&klds[0][sr * KSTR + sc0]     = pk8(k0, k1);
    *(v8h*)&klds[0][sr * KSTR + sc0 + 8] = pk8(k2, k3);
    v4f f0 = *(const v4f*)gv, f1 = *(const v4f*)(gv + 4);
    v4f f2 = *(const v4f*)(gv + 8), f3 = *(const v4f*)(gv + 12);
#pragma unroll
    for (int i = 0; i < 4; ++i) {
      vtlds[0][(sc0 + i) * VSTR + sp]      = (_Float16)f0[i];
      vtlds[0][(sc0 + 4 + i) * VSTR + sp]  = (_Float16)f1[i];
      vtlds[0][(sc0 + 8 + i) * VSTR + sp]  = (_Float16)f2[i];
      vtlds[0][(sc0 + 12 + i) * VSTR + sp] = (_Float16)f3[i];
    }
  }
  __syncthreads();

  // ---- per-wave Q A-fragments (contiguous k = 16*lh + s layout) ----
  const int qrow = 16 * wv + ln;
  const _Float16* qr = &qlds[qrow * QSTR + 16 * lh];
  v16h Qa0 = cat16(*(const v8h*)qr, *(const v8h*)(qr + 8));          // E 0..31
  v16h Qa1 = cat16(*(const v8h*)(qr + 32), *(const v8h*)(qr + 40));  // E 32..63

  v16h ones = {(_Float16)1.0f, (_Float16)1.0f, (_Float16)1.0f, (_Float16)1.0f,
               (_Float16)1.0f, (_Float16)1.0f, (_Float16)1.0f, (_Float16)1.0f,
               (_Float16)1.0f, (_Float16)1.0f, (_Float16)1.0f, (_Float16)1.0f,
               (_Float16)1.0f, (_Float16)1.0f, (_Float16)1.0f, (_Float16)1.0f};

  v8f acc0 = {}, acc1 = {}, acc2 = {}, acc3 = {};
  v8f accl = {};  // softmax denominator, tracked by the P x ones WMMA
  float m[8];
#pragma unroll
  for (int r = 0; r < 8; ++r) m[r] = -1e30f;

  _Float16* pbase = &plds[wv * 16 * PSTR];

  int pb = 0;
  for (int j0 = 0; j0 < L_; j0 += 32) {
    // ---- issue global loads for the NEXT tile (hidden behind WMMA work) ----
    Stage st;
    {
      int jn = (j0 + 32 < L_) ? (j0 + 32) : j0;  // clamp on last iteration
      const float* gk = kbase + (size_t)(jn + sr) * rowstride + sc0;
      const float* gv = vbase + (size_t)(jn + sr) * rowstride + sc0;
      st.k0 = *(const v4f*)gk;       st.k1 = *(const v4f*)(gk + 4);
      st.k2 = *(const v4f*)(gk + 8); st.k3 = *(const v4f*)(gk + 12);
      st.v0 = *(const v4f*)gv;       st.v1 = *(const v4f*)(gv + 4);
      st.v2 = *(const v4f*)(gv + 8); st.v3 = *(const v4f*)(gv + 12);
    }

    const _Float16* kl = klds[pb];
    const _Float16* vl = vtlds[pb];

    // ---- scores: two 16x16 tiles, already scaled to the exp2 domain ----
    v8f s0, s1;
    {
      v8f z = {};
      const _Float16* ka = &kl[ln * KSTR + 16 * lh];        // keys +0..15, E 0..31
      s0 = wmma16(Qa0, cat16(*(const v8h*)ka, *(const v8h*)(ka + 8)), z);
      const _Float16* kb = &kl[ln * KSTR + 32 + 16 * lh];   // E 32..63
      s0 = wmma16(Qa1, cat16(*(const v8h*)kb, *(const v8h*)(kb + 8)), s0);
      const _Float16* kc = &kl[(16 + ln) * KSTR + 16 * lh]; // keys +16..31
      s1 = wmma16(Qa0, cat16(*(const v8h*)kc, *(const v8h*)(kc + 8)), z);
      const _Float16* kd = &kl[(16 + ln) * KSTR + 32 + 16 * lh];
      s1 = wmma16(Qa1, cat16(*(const v8h*)kd, *(const v8h*)(kd + 8)), s1);
    }

    // ---- online softmax: row r+8*lh lives in vgpr r, striped over 16 lanes ----
#pragma unroll
    for (int r = 0; r < 8; ++r) {
      float a  = s0[r];
      float bb = s1[r];
      float mn = rawmax(m[r], rowmax16(rawmax(a, bb)));
      float al = fexp2(m[r] - mn);
      float p0 = fexp2(a - mn);
      float p1 = fexp2(bb - mn);
      m[r] = mn;
      // packed P spill: columns of the two tiles interleaved (key-permuted)
      *(v2h*)&pbase[(r + 8 * lh) * PSTR + 2 * ln] =
          __builtin_bit_cast(v2h, __builtin_amdgcn_cvt_pkrtz(p0, p1));
      acc0[r] *= al;
      acc1[r] *= al;
      acc2[r] *= al;
      acc3[r] *= al;
      accl[r] *= al;
    }

    // ---- reload P as A-fragment (16 rows x 32 permuted keys) ----
    const _Float16* pr = &pbase[ln * PSTR + 16 * lh];
    v16h Pa = cat16(*(const v8h*)pr, *(const v8h*)(pr + 8));

    // ---- O += P x V ; l += P x ones (row-sum via WMMA, lane-replicated) ----
    {
      const _Float16* v0 = &vl[ln * VSTR + 16 * lh];
      acc0 = wmma16(Pa, cat16(*(const v8h*)v0, *(const v8h*)(v0 + 8)), acc0);
      const _Float16* v1 = &vl[(16 + ln) * VSTR + 16 * lh];
      acc1 = wmma16(Pa, cat16(*(const v8h*)v1, *(const v8h*)(v1 + 8)), acc1);
      const _Float16* v2 = &vl[(32 + ln) * VSTR + 16 * lh];
      acc2 = wmma16(Pa, cat16(*(const v8h*)v2, *(const v8h*)(v2 + 8)), acc2);
      const _Float16* v3 = &vl[(48 + ln) * VSTR + 16 * lh];
      acc3 = wmma16(Pa, cat16(*(const v8h*)v3, *(const v8h*)(v3 + 8)), acc3);
      accl = wmma16(Pa, ones, accl);
    }

    // ---- store staged next tile into the other buffer, flip ----
    {
      _Float16* kn = (_Float16*)klds[1 - pb];
      _Float16* vn = (_Float16*)vtlds[1 - pb];
      *(v8h*)&kn[sr * KSTR + sc0]     = pk8(st.k0, st.k1);
      *(v8h*)&kn[sr * KSTR + sc0 + 8] = pk8(st.k2, st.k3);
#pragma unroll
      for (int i = 0; i < 4; ++i) {
        vn[(sc0 + i) * VSTR + sp]      = (_Float16)st.v0[i];
        vn[(sc0 + 4 + i) * VSTR + sp]  = (_Float16)st.v1[i];
        vn[(sc0 + 8 + i) * VSTR + sp]  = (_Float16)st.v2[i];
        vn[(sc0 + 12 + i) * VSTR + sp] = (_Float16)st.v3[i];
      }
    }
    __syncthreads();
    pb ^= 1;
  }

  // ---- epilogue: l is lane-replicated in accl; normalize and store ----
#pragma unroll
  for (int r = 0; r < 8; ++r) {
    float inv = 1.0f / accl[r];
    int grow = q0 + 16 * wv + r + 8 * lh;
    float* o = obase + (size_t)grow * rowstride;
    o[ln]      = acc0[r] * inv;
    o[16 + ln] = acc1[r] * inv;
    o[32 + ln] = acc2[r] * inv;
    o[48 + ln] = acc3[r] * inv;
  }
}

extern "C" void kernel_launch(void* const* d_in, const int* in_sizes, int n_in,
                              void* d_out, int out_size, void* d_ws, size_t ws_size,
                              hipStream_t stream) {
  const float* Q = (const float*)d_in[0];
  const float* K = (const float*)d_in[1];
  const float* V = (const float*)d_in[2];
  float* O = (float*)d_out;
  dim3 grid(L_ / 64, B_ * H_);  // 32 q-tiles x 32 (b,h)
  flash_attn_f16wmma<<<grid, 128, 0, stream>>>(Q, K, V, O);
}
